// StructSeqEnc_35811437314203
// MI455X (gfx1250) — matrix operand order repaired and checked by reading, not compiled, not run on hardware
//
#include <hip/hip_runtime.h>

typedef unsigned int  u32;
typedef unsigned short u16;
typedef __bf16 v16bf __attribute__((ext_vector_type(16)));
typedef float  v8f   __attribute__((ext_vector_type(8)));

#define ASYNC_CP16(ldsaddr, base, voff)                                        \
  asm volatile("global_load_async_to_lds_b128 %0, %1, %2"                      \
               :: "v"(ldsaddr), "v"(voff), "s"(base) : "memory")

__device__ __forceinline__ u32 bf16_rne(float x) {
  u32 u = __float_as_uint(x);
  return (u + 0x7FFFu + ((u >> 16) & 1u)) >> 16;
}
// hi = truncated-bf16(x), lo = rne-bf16(x - hi): x ~= hi + lo to ~2^-17 rel.
__device__ __forceinline__ void split1(float x, u32& h, u32& l) {
  u32 u = __float_as_uint(x);
  h = u >> 16;
  float r = x - __uint_as_float(u & 0xFFFF0000u);
  l = bf16_rne(r);
}

__device__ __forceinline__ v8f wmma_bf16(v16bf a, v16bf b, v8f c) {
  return __builtin_amdgcn_wmma_f32_16x16x32_bf16(false, a, false, b,
                                                 (short)0, c, false, false);
}
// A fragment (16x32 bf16): chunk0 at p, chunk1 at p+32 bytes
__device__ __forceinline__ v16bf ldA(const char* p) {
  union { uint4 q[2]; v16bf v; } u;
  u.q[0] = *(const uint4*)p;
  u.q[1] = *(const uint4*)(p + 32);
  return u.v;
}
// B fragment (32x16 bf16): chunk0 at p, chunk1 at p+32 bytes
__device__ __forceinline__ v16bf ldB(const char* p) {
  union { uint4 q[2]; v16bf v; } u;
  u.q[0] = *(const uint4*)p;
  u.q[1] = *(const uint4*)(p + 32);
  return u.v;
}

// ---------------- pre-pass: split fp32 -> bf16 hi/lo (row-major) -------------
__global__ __launch_bounds__(256) void k_split(const float* __restrict__ src,
                                               u16* __restrict__ dH,
                                               u16* __restrict__ dL, int n4) {
  int stride = gridDim.x * blockDim.x;
  for (int i = blockIdx.x * blockDim.x + threadIdx.x; i < n4; i += stride) {
    float4 v = ((const float4*)src)[i];
    u32 h0,l0,h1,l1,h2,l2,h3,l3;
    split1(v.x,h0,l0); split1(v.y,h1,l1); split1(v.z,h2,l2); split1(v.w,h3,l3);
    ((uint2*)dH)[i] = make_uint2(h0 | (h1 << 16), h2 | (h3 << 16));
    ((uint2*)dL)[i] = make_uint2(l0 | (l1 << 16), l2 | (l3 << 16));
  }
}

// ------------- pre-pass: split + transpose W -> Wt[col][k] bf16 hi/lo --------
__global__ __launch_bounds__(256) void k_splitT(const float* __restrict__ W,
                                                u16* __restrict__ tH,
                                                u16* __restrict__ tL) {
  __shared__ u16 sH[64][66];
  __shared__ u16 sL[64][66];
  int tid = threadIdx.x;
  int k0 = blockIdx.y * 64, c0 = blockIdx.x * 64;
#pragma unroll
  for (int it = 0; it < 16; ++it) {
    int e = tid + it * 256, r = e >> 6, c = e & 63;
    u32 h, l;
    split1(W[(size_t)(k0 + r) * 1024 + c0 + c], h, l);
    sH[r][c] = (u16)h; sL[r][c] = (u16)l;
  }
  __syncthreads();
#pragma unroll
  for (int it = 0; it < 8; ++it) {
    int e = tid + it * 256, c = e >> 5, k = (e & 31) * 2;
    u32 H = (u32)sH[k][c] | ((u32)sH[k + 1][c] << 16);
    u32 L = (u32)sL[k][c] | ((u32)sL[k + 1][c] << 16);
    size_t o = (size_t)(c0 + c) * 1024 + k0 + k;
    *(u32*)(tH + o) = H;
    *(u32*)(tL + o) = L;
  }
}

// ---------------- fused dual-GEMM -> rowwise dot (h1) ------------------------
// LDS stage buffer layout (bytes), pitch 80 per 32-k row:
//  gabH 0 / gabL 5120 / gagH 10240 / gagL 15360  (64 rows x 64B)
//  wabH 20480 / wabL 30720 / wagH 40960 / wagL 51200 (128 cols x 64B)
//  SSZ = 61440 per buffer; bias @ 122880 (8KB); sH @ 131072 (256B)
__global__ __launch_bounds__(256) void k_h1(
    const u16* __restrict__ gabH, const u16* __restrict__ gabL,
    const u16* __restrict__ gagH, const u16* __restrict__ gagL,
    const u16* __restrict__ wabH, const u16* __restrict__ wabL,
    const u16* __restrict__ wagH, const u16* __restrict__ wagL,
    const float* __restrict__ bab, const float* __restrict__ bag,
    float* __restrict__ h1out) {
  extern __shared__ __align__(16) char smem[];
  const int tid  = threadIdx.x;
  const int lane = tid & 31;
  const int wid  = tid >> 5;
  const int rt = wid & 3, jc = wid >> 2;
  const int rowBase = blockIdx.x * 64;
  const u32 smemA = (u32)(size_t)smem;
  float* sHf = (float*)(smem + 131072);
  const float* sb = (const float*)(smem + 122880);
  if (tid < 64) sHf[tid] = 0.0f;

  // biases -> LDS via async copy (256 threads x 16B covers 4KB each)
  {
    u32 lb  = smemA + 122880u + (u32)tid * 16u;
    u32 off = (u32)tid * 16u;
    ASYNC_CP16(lb, bab, off);
    ASYNC_CP16(lb + 4096u, bag, off);
  }

  auto issue = [&](int s, int b) {
    int jg = s >> 5;
    u32 k0b = (u32)(s & 31) * 64u;          // 32 k-elements = 64 bytes
    u32 bb  = smemA + (u32)b * 61440u;
    u32 r4  = (u32)(tid >> 2);
    u32 ck  = (u32)(tid & 3) * 16u;
    u32 gl  = bb + r4 * 80u + ck;
    u32 gs  = (u32)(rowBase + (int)r4) * 2048u + k0b + ck;
    ASYNC_CP16(gl,           gabH, gs);
    ASYNC_CP16(gl + 5120u,   gabL, gs);
    ASYNC_CP16(gl + 10240u,  gagH, gs);
    ASYNC_CP16(gl + 15360u,  gagL, gs);
    u32 jgb = (u32)jg * 128u;
#pragma unroll
    for (int half = 0; half < 2; ++half) {
      u32 c  = (u32)half * 64u + r4;
      u32 wl = bb + 20480u + c * 80u + ck;
      u32 wo = (jgb + c) * 2048u + k0b + ck;
      ASYNC_CP16(wl,           wabH, wo);
      ASYNC_CP16(wl + 10240u,  wabL, wo);
      ASYNC_CP16(wl + 20480u,  wagH, wo);
      ASYNC_CP16(wl + 30720u,  wagL, wo);
    }
  };

  v8f hp = {0,0,0,0,0,0,0,0};
  v8f accAB[4], accAG[4];
  const u32 row  = (u32)(rt * 16 + (lane & 15));
  const u32 aSel = (u32)((lane >> 4) << 4);   // A: +0 / +16 bytes
  const u32 bSel = (u32)((lane >> 4) << 4);   // B: +0 / +16 bytes

  // one pipeline step: wait stage s, prefetch stage s+1, 24 WMMAs on buf
  auto step = [&](int s, int buf) {
    asm volatile("s_wait_asynccnt 0x0" ::: "memory");
    __syncthreads();
    if (s + 1 < 256) issue(s + 1, (s + 1) & 1);   // overlap copy with compute
    const char* bb = smem + (size_t)buf * 61440;
    const char* ap = bb + row * 80u + aSel;
    v16bf aabH = ldA(ap);
    v16bf aabL = ldA(ap + 5120);
    v16bf aagH = ldA(ap + 10240);
    v16bf aagL = ldA(ap + 15360);
#pragma unroll
    for (int jt = 0; jt < 4; ++jt) {
      u32 c = (u32)(jc * 64 + jt * 16 + (lane & 15));
      const char* bp = bb + 20480u + c * 80u + bSel;
      v16bf bH = ldB(bp);
      v16bf bL = ldB(bp + 10240);
      v16bf cH = ldB(bp + 20480);
      v16bf cL = ldB(bp + 30720);
      accAB[jt] = wmma_bf16(aabH, bH, accAB[jt]);   // hi*hi
      accAB[jt] = wmma_bf16(aabH, bL, accAB[jt]);   // hi*lo
      accAB[jt] = wmma_bf16(aabL, bH, accAB[jt]);   // lo*hi
      accAG[jt] = wmma_bf16(aagH, cH, accAG[jt]);
      accAG[jt] = wmma_bf16(aagH, cL, accAG[jt]);
      accAG[jt] = wmma_bf16(aagL, cH, accAG[jt]);
    }
  };

  issue(0, 0);
  for (int jg = 0; jg < 8; ++jg) {
    {
      v8f z = {0,0,0,0,0,0,0,0};
#pragma unroll
      for (int jt = 0; jt < 4; ++jt) { accAB[jt] = z; accAG[jt] = z; }
    }
    for (int k0i = 0; k0i < 32; k0i += 2) {   // static double-buffer parity
      int s = jg * 32 + k0i;
      step(s, 0);
      step(s + 1, 1);
    }
    // j-group finished: add bias, multiply encoders, accumulate row partials
#pragma unroll
    for (int jt = 0; jt < 4; ++jt) {
      int col = jg * 128 + jc * 64 + jt * 16 + (lane & 15);
      float bA = sb[col], bG = sb[1024 + col];
#pragma unroll
      for (int v = 0; v < 8; ++v)
        hp[v] += (accAB[jt][v] + bA) * (accAG[jt][v] + bG);
    }
  }
  // C layout: VGPR v, lanes 0-15 -> row v, lanes 16-31 -> row v+8
#pragma unroll
  for (int v = 0; v < 8; ++v) {
    float x = hp[v];
    x += __shfl_xor(x, 1); x += __shfl_xor(x, 2);
    x += __shfl_xor(x, 4); x += __shfl_xor(x, 8);
    if ((lane & 15) == 0)
      atomicAdd(&sHf[rt * 16 + v + ((lane >> 4) << 3)], x);
  }
  __syncthreads();
  if (tid < 64) h1out[rowBase + tid] = sHf[tid];
}

// ---------------- streaming: h2 = rowdot(sAB,sAG); out = (h1+h2)/2 -----------
__global__ __launch_bounds__(256) void k_h2(const float* __restrict__ sAB,
                                            const float* __restrict__ sAG,
                                            const float* __restrict__ h1,
                                            float* __restrict__ out) {
  int lane = threadIdx.x & 31;
  int row  = blockIdx.x * 8 + (threadIdx.x >> 5);
  const float4* a = (const float4*)(sAB + (size_t)row * 1024);
  const float4* b = (const float4*)(sAG + (size_t)row * 1024);
  float acc = 0.f;
#pragma unroll
  for (int i = 0; i < 8; ++i) {
    float4 x = a[lane + i * 32], y = b[lane + i * 32];
    acc += x.x * y.x + x.y * y.y + x.z * y.z + x.w * y.w;
  }
  acc += __shfl_xor(acc, 16); acc += __shfl_xor(acc, 8);
  acc += __shfl_xor(acc, 4);  acc += __shfl_xor(acc, 2);
  acc += __shfl_xor(acc, 1);
  if (lane == 0) out[row] = 0.5f * (h1[row] + acc);
}

extern "C" void kernel_launch(void* const* d_in, const int* in_sizes, int n_in,
                              void* d_out, int out_size, void* d_ws, size_t ws_size,
                              hipStream_t stream) {
  (void)in_sizes; (void)n_in; (void)out_size; (void)ws_size;
  const float* gAB = (const float*)d_in[0];
  const float* gAG = (const float*)d_in[1];
  const float* Wab = (const float*)d_in[2];
  const float* Wag = (const float*)d_in[3];
  const float* bab = (const float*)d_in[4];
  const float* bag = (const float*)d_in[5];
  const float* sAB = (const float*)d_in[6];
  const float* sAG = (const float*)d_in[7];

  char* ws = (char*)d_ws;
  const size_t G   = (size_t)8192 * 1024 * 2;  // 16 MB per g half-matrix
  const size_t WSZ = (size_t)1024 * 1024 * 2;  //  2 MB per W half-matrix
  u16* gabH = (u16*)(ws);
  u16* gabL = (u16*)(ws + G);
  u16* gagH = (u16*)(ws + 2 * G);
  u16* gagL = (u16*)(ws + 3 * G);
  u16* wabH = (u16*)(ws + 4 * G);
  u16* wabL = (u16*)(ws + 4 * G + WSZ);
  u16* wagH = (u16*)(ws + 4 * G + 2 * WSZ);
  u16* wagL = (u16*)(ws + 4 * G + 3 * WSZ);
  float* h1 = (float*)(ws + 4 * G + 4 * WSZ);

  int n4 = 8192 * 1024 / 4;
  k_split<<<2048, 256, 0, stream>>>(gAB, gabH, gabL, n4);
  k_split<<<2048, 256, 0, stream>>>(gAG, gagH, gagL, n4);
  dim3 gT(16, 16);
  k_splitT<<<gT, 256, 0, stream>>>(Wab, wabH, wabL);
  k_splitT<<<gT, 256, 0, stream>>>(Wag, wagH, wagL);
  k_h1<<<128, 256, 131328, stream>>>(gabH, gabL, gagH, gagL,
                                     wabH, wabL, wagH, wagL, bab, bag, h1);
  k_h2<<<1024, 256, 0, stream>>>(sAB, sAG, h1, (float*)d_out);
}
